// TransformerXL_21500606284390
// MI455X (gfx1250) — matrix operand order repaired and checked by reading, not compiled, not run on hardware
//
#include <hip/hip_runtime.h>
#include <hip/hip_bf16.h>
#include <stdint.h>

// ---------------- problem constants ----------------
#define VOCABN 32000
#define Dm     512
#define Hh     8
#define DKn    64
#define Ln     2
#define FFn    2048
#define MREL   512
#define Bn     4
#define Sn     512
#define NREL   (2*MREL+1)   // 1025
#define BS     (Bn*Sn)      // 2048

typedef unsigned short u16;
typedef __attribute__((ext_vector_type(16))) __bf16 v16bf;
typedef __attribute__((ext_vector_type(8)))  float  v8f;
typedef __attribute__((ext_vector_type(4)))  int    v4i;

__device__ __forceinline__ u16 f2bf(float f) {
  unsigned u = __float_as_uint(f);
  u += 0x7FFFu + ((u >> 16) & 1u);        // round-to-nearest-even
  return (u16)(u >> 16);
}

// ---------------- gfx1250 async global->LDS copy (128-bit per lane) ----------------
#if __has_builtin(__builtin_amdgcn_global_load_async_to_lds_b128)
#define ASYNC_BUILTIN 1
#else
#define ASYNC_BUILTIN 0
#endif

typedef __attribute__((address_space(1))) v4i gas_v4i;   // global
typedef __attribute__((address_space(3))) v4i las_v4i;   // LDS

__device__ __forceinline__ void async_cp16(const u16* g, u16* l) {
#if ASYNC_BUILTIN
  __builtin_amdgcn_global_load_async_to_lds_b128((gas_v4i*)g, (las_v4i*)l, 0, 0);
#else
  // VGLOBAL async encoding: VDST = per-lane LDS byte address, VADDR = 64-bit global addr
  unsigned lds_off = (unsigned)(uintptr_t)l;     // low 32 bits of generic shared ptr = LDS offset
  unsigned long long ga = (unsigned long long)(uintptr_t)g;
  asm volatile("global_load_async_to_lds_b128 %0, %1, off"
               :: "v"(lds_off), "v"(ga) : "memory");
#endif
}

__device__ __forceinline__ void wait_async0() {
#if __has_builtin(__builtin_amdgcn_s_wait_asynccnt)
  __builtin_amdgcn_s_wait_asynccnt(0);
#else
  asm volatile("s_wait_asynccnt 0x0" ::: "memory");
#endif
}

// ---------------- elementwise kernels ----------------
__global__ void k_f32_to_bf16(const float* __restrict__ src, u16* __restrict__ dst, int n) {
  int i = blockIdx.x * blockDim.x + threadIdx.x;
  if (i < n) dst[i] = f2bf(src[i]);
}

__global__ void k_embed(const int* __restrict__ ids, const float* __restrict__ emb,
                        float* __restrict__ h32) {
  int row = blockIdx.x;                   // 0..BS-1
  long id = (long)ids[row];
  const float* e = emb + id * Dm;
  float* o = h32 + (long)row * Dm;
  for (int t = threadIdx.x; t < Dm; t += blockDim.x)
    o[t] = e[t] * 22.627416997969522f;    // sqrt(512)
}

// qc = bf16(q + u[d]), qp = bf16(q + v[d]); u/v already offset to layer l
__global__ void k_bias_head2(const float* __restrict__ q32, const float* __restrict__ u,
                             const float* __restrict__ v, u16* __restrict__ qc,
                             u16* __restrict__ qp) {
  int i = blockIdx.x * blockDim.x + threadIdx.x;
  if (i >= BS * Dm) return;
  int d = i & (Dm - 1);
  float q = q32[i];
  qc[i] = f2bf(q + u[d]);
  qp[i] = f2bf(q + v[d]);
}

// ---------------- WMMA GEMM ----------------
// C[M,N](f32 and/or bf16) = act(A[M,K]_bf16 * B_bf16 + bias)
// BT=0: B is K x N row-major.  BT=1: B is N x K row-major (C = A * B^T).
// Block tile 128(M) x 64(N), 128 threads = 4 waves, each wave owns 32 rows.
// Double-buffered LDS staging via async global->LDS copies (ASYNCcnt).
__launch_bounds__(128)
__global__ void k_gemm(const u16* __restrict__ A, const u16* __restrict__ B,
                       float* __restrict__ C, u16* __restrict__ Cb,
                       const float* __restrict__ bias,
                       int M, int N, int K, int lda, int ldb, int ldc,
                       long sAb, long sAh, long sBb, long sBh, long sCb, long sCh,
                       int HS, int BT, int relu) {
  __shared__ u16 lsA[2][128][40];   // [buf][m][k], row stride 80B (16B-aligned chunks)
  __shared__ u16 lsB[2][64][40];    // [buf][n][k] (B held transposed in LDS)

  int z = blockIdx.z;
  int zb = z / HS, zh = z % HS;
  A += zb * sAb + (long)zh * sAh;
  B += zb * sBb + (long)zh * sBh;
  long coff = zb * sCb + (long)zh * sCh;

  int n0 = blockIdx.x * 64;
  int m0 = blockIdx.y * 128;
  int tid  = threadIdx.x;
  int wave = tid >> 5, lane = tid & 31;
  int hi   = lane >> 4, ml = lane & 15;

  v8f acc[2][4];
#pragma unroll
  for (int i = 0; i < 2; ++i)
#pragma unroll
    for (int j = 0; j < 4; ++j)
#pragma unroll
      for (int e = 0; e < 8; ++e) acc[i][j][e] = 0.0f;

  union FragU { uint4 u4[2]; v16bf v; };

  // ---- staging: A + (BT) B via async copies; rr-mode B via sync transpose ----
  auto stage = [&](int buf, int k0) {
    // A tile: 128 x 32, one row per thread, 4 async 16B chunks
    {
      int gm = m0 + tid;
      if (gm < M) {
        const u16* src = A + (long)gm * lda + k0;
        u16* dst = &lsA[buf][tid][0];
#pragma unroll
        for (int c = 0; c < 4; ++c) async_cp16(src + c * 8, dst + c * 8);
      }
    }
    if (BT) {
      // B tile (N x K source): 64 rows, 2 async 16B chunks per thread
      int r = tid >> 1, half = tid & 1;
      int gn = n0 + r;
      if (gn < N) {
        const u16* src = B + (long)gn * ldb + k0 + half * 16;
        u16* dst = &lsB[buf][r][half * 16];
        async_cp16(src, dst);
        async_cp16(src + 8, dst + 8);
      }
    } else {
      // B tile (K x N source): transpose into [n][k] synchronously
      int kr = tid >> 2, q = tid & 3;           // kr: 0..31, q: 0..3
      int nb = n0 + q * 16;
      const u16* p = B + (long)(k0 + kr) * ldb + nb;
      if (k0 + 32 < K) __builtin_prefetch(p + (long)32 * ldb, 0, 3);
      u16 tmp[16];
#pragma unroll
      for (int i = 0; i < 16; ++i) tmp[i] = (nb + i < N) ? p[i] : (u16)0;
#pragma unroll
      for (int i = 0; i < 16; ++i) lsB[buf][q * 16 + i][kr] = tmp[i];
    }
  };

  int nk = K >> 5;                 // K is always a multiple of 32 here
  stage(0, 0);
  for (int it = 0; it < nk; ++it) {
    int cur = it & 1;
    wait_async0();                 // this wave's async copies have landed
    __syncthreads();               // all waves' tiles (async + ds_store) visible
    if (it + 1 < nk) stage(1 - cur, (it + 1) << 5);

    // ---- fragments (ISA 7.12.2 bf16 layouts) ----
    FragU fa0, fa1;
    int mrow = wave * 32 + ml;
    fa0.u4[0] = *(const uint4*)&lsA[cur][mrow][hi * 8];
    fa0.u4[1] = *(const uint4*)&lsA[cur][mrow][16 + hi * 8];
    fa1.u4[0] = *(const uint4*)&lsA[cur][mrow + 16][hi * 8];
    fa1.u4[1] = *(const uint4*)&lsA[cur][mrow + 16][16 + hi * 8];
#pragma unroll
    for (int nt = 0; nt < 4; ++nt) {
      FragU fb;
      fb.u4[0] = *(const uint4*)&lsB[cur][nt * 16 + ml][hi * 8];
      fb.u4[1] = *(const uint4*)&lsB[cur][nt * 16 + ml][16 + hi * 8];
      acc[0][nt] = __builtin_amdgcn_wmma_f32_16x16x32_bf16(
          false, fa0.v, false, fb.v, (short)0, acc[0][nt], false, false);
      acc[1][nt] = __builtin_amdgcn_wmma_f32_16x16x32_bf16(
          false, fa1.v, false, fb.v, (short)0, acc[1][nt], false, false);
    }
    __syncthreads();               // everyone done reading buf `cur`
  }

  // ---- epilogue ----
#pragma unroll
  for (int mt = 0; mt < 2; ++mt) {
#pragma unroll
    for (int nt = 0; nt < 4; ++nt) {
      int col = n0 + nt * 16 + ml;
      if (col >= N) continue;
      float bsv = bias ? bias[col] : 0.0f;
#pragma unroll
      for (int r = 0; r < 8; ++r) {
        int row = m0 + wave * 32 + mt * 16 + hi * 8 + r;
        if (row < M) {
          float val = acc[mt][nt][r] + bsv;
          if (relu) val = val > 0.0f ? val : 0.0f;
          long idx = coff + (long)row * ldc + col;
          if (C)  C[idx]  = val;
          if (Cb) Cb[idx] = f2bf(val);
        }
      }
    }
  }
}

// ---------------- softmax with relative shift + causal mask ----------------
__launch_bounds__(128)
__global__ void k_softmax(const float* __restrict__ content, const float* __restrict__ posf,
                          u16* __restrict__ attn) {
  int bid = blockIdx.x;
  int i = bid & (Sn - 1);
  const float* cr = content + (long)bid * Sn;
  const float* pr = posf + (long)bid * NREL;
  u16* ar = attn + (long)bid * Sn;
  const float scale = 0.125f;             // 1/sqrt(64)
  int tid = threadIdx.x;
  float vals[4];
  float mx = -1e30f;
#pragma unroll
  for (int e = 0; e < 4; ++e) {
    int j = tid + e * 128;
    int t = i - j + MREL;                 // in [1,1023]
    float v = (j <= i) ? (cr[j] + pr[t]) * scale : -1e9f;
    vals[e] = v;
    mx = fmaxf(mx, v);
  }
  __shared__ float red[128];
  red[tid] = mx; __syncthreads();
  for (int s = 64; s > 0; s >>= 1) { if (tid < s) red[tid] = fmaxf(red[tid], red[tid + s]); __syncthreads(); }
  mx = red[0]; __syncthreads();
  float sum = 0.0f;
#pragma unroll
  for (int e = 0; e < 4; ++e) { vals[e] = __expf(vals[e] - mx); sum += vals[e]; }
  red[tid] = sum; __syncthreads();
  for (int s = 64; s > 0; s >>= 1) { if (tid < s) red[tid] += red[tid + s]; __syncthreads(); }
  float inv = 1.0f / red[0];
#pragma unroll
  for (int e = 0; e < 4; ++e) ar[tid + e * 128] = f2bf(vals[e] * inv);
}

// ---------------- residual + layernorm (in place on h32) ----------------
__launch_bounds__(128)
__global__ void k_addln(float* __restrict__ h, const float* __restrict__ delta,
                        const float* __restrict__ g, const float* __restrict__ be) {
  int row = blockIdx.x;
  float* hr = h + (long)row * Dm;
  const float* dr = delta + (long)row * Dm;
  int tid = threadIdx.x;
  float x[4];
  float s = 0.0f;
#pragma unroll
  for (int e = 0; e < 4; ++e) { int c = tid + e * 128; x[e] = hr[c] + dr[c]; s += x[e]; }
  __shared__ float red[128];
  red[tid] = s; __syncthreads();
  for (int st = 64; st > 0; st >>= 1) { if (tid < st) red[tid] += red[tid + st]; __syncthreads(); }
  float mean = red[0] * (1.0f / Dm); __syncthreads();
  float vs = 0.0f;
#pragma unroll
  for (int e = 0; e < 4; ++e) { float d0 = x[e] - mean; vs += d0 * d0; }
  red[tid] = vs; __syncthreads();
  for (int st = 64; st > 0; st >>= 1) { if (tid < st) red[tid] += red[tid + st]; __syncthreads(); }
  float rinv = rsqrtf(red[0] * (1.0f / Dm) + 1e-5f);
#pragma unroll
  for (int e = 0; e < 4; ++e) { int c = tid + e * 128; hr[c] = (x[e] - mean) * rinv * g[c] + be[c]; }
}

// ---------------- host-side helpers ----------------
static inline void conv_launch(hipStream_t s, const float* src, u16* dst, long n) {
  k_f32_to_bf16<<<dim3((unsigned)((n + 255) / 256)), dim3(256), 0, s>>>(src, dst, (int)n);
}

static inline void gemm_launch(hipStream_t s, const u16* A, const u16* B, float* C, u16* Cb,
                               const float* bias, int M, int N, int K,
                               int lda, int ldb, int ldc,
                               long sAb, long sAh, long sBb, long sBh, long sCb, long sCh,
                               int HS, int nbatch, int BT, int relu) {
  dim3 grid((N + 63) / 64, (M + 127) / 128, nbatch);
  k_gemm<<<grid, dim3(128), 0, s>>>(A, B, C, Cb, bias, M, N, K, lda, ldb, ldc,
                                    sAb, sAh, sBb, sBh, sCb, sCh, HS, BT, relu);
}

extern "C" void kernel_launch(void* const* d_in, const int* in_sizes, int n_in,
                              void* d_out, int out_size, void* d_ws, size_t ws_size,
                              hipStream_t stream) {
  (void)in_sizes; (void)n_in; (void)out_size; (void)ws_size;
  const int*   ids   = (const int*)  d_in[0];
  const float* emb   = (const float*)d_in[1];
  const float* rel   = (const float*)d_in[2];
  const float* Wq    = (const float*)d_in[3];
  const float* Wk    = (const float*)d_in[4];
  const float* Wv    = (const float*)d_in[5];
  const float* Wo    = (const float*)d_in[6];
  const float* Wr    = (const float*)d_in[7];
  const float* uB    = (const float*)d_in[8];
  const float* vB    = (const float*)d_in[9];
  const float* W1    = (const float*)d_in[10];
  const float* b1    = (const float*)d_in[11];
  const float* W2    = (const float*)d_in[12];
  const float* b2    = (const float*)d_in[13];
  const float* g1    = (const float*)d_in[14];
  const float* be1   = (const float*)d_in[15];
  const float* g2    = (const float*)d_in[16];
  const float* be2   = (const float*)d_in[17];
  const float* Wout  = (const float*)d_in[18];
  const float* bout  = (const float*)d_in[19];
  float* out = (float*)d_out;

  // bump allocator on workspace
  uint8_t* wsp = (uint8_t*)d_ws;
  size_t off = 0;
  auto alloc = [&](size_t bytes) -> void* {
    void* p = wsp + off;
    off = (off + bytes + 255) & ~(size_t)255;
    return p;
  };
  float* h32     = (float*)alloc((size_t)BS * Dm * 4);
  float* tmp32   = (float*)alloc((size_t)BS * Dm * 4);       // also holds 1025x512 r-proj
  float* d32     = (float*)alloc((size_t)BS * Dm * 4);
  float* o32     = (float*)alloc((size_t)BS * Dm * 4);
  u16*   hb      = (u16*)  alloc((size_t)BS * Dm * 2);
  u16*   qc      = (u16*)  alloc((size_t)BS * Dm * 2);
  u16*   qp      = (u16*)  alloc((size_t)BS * Dm * 2);
  u16*   kb      = (u16*)  alloc((size_t)BS * Dm * 2);
  u16*   vb      = (u16*)  alloc((size_t)BS * Dm * 2);
  u16*   ob      = (u16*)  alloc((size_t)BS * Dm * 2);
  u16*   rb      = (u16*)  alloc((size_t)NREL * Dm * 2);
  u16*   relb    = (u16*)  alloc((size_t)NREL * Dm * 2);
  float* content = (float*)alloc((size_t)Bn * Hh * Sn * Sn * 4);    // 32 MB
  float* posf    = (float*)alloc((size_t)Bn * Hh * Sn * NREL * 4);  // 67 MB
  u16*   attn_b  = (u16*)  alloc((size_t)Bn * Hh * Sn * Sn * 2);    // 16 MB
  u16*   f1b     = (u16*)  alloc((size_t)BS * FFn * 2);             // 8 MB
  u16*   wqb     = (u16*)  alloc((size_t)Dm * Dm * 2);
  u16*   wkb     = (u16*)  alloc((size_t)Dm * Dm * 2);
  u16*   wvb     = (u16*)  alloc((size_t)Dm * Dm * 2);
  u16*   wob     = (u16*)  alloc((size_t)Dm * Dm * 2);
  u16*   wrb     = (u16*)  alloc((size_t)Dm * Dm * 2);
  u16*   w1b     = (u16*)  alloc((size_t)Dm * FFn * 2);
  u16*   w2b     = (u16*)  alloc((size_t)FFn * Dm * 2);
  u16*   woutb   = (u16*)  alloc((size_t)Dm * VOCABN * 2);          // 32.8 MB

  const long SD = (long)Sn * Dm;

  // one-time conversions
  conv_launch(stream, rel,  relb,  (long)NREL * Dm);
  conv_launch(stream, Wout, woutb, (long)Dm * VOCABN);

  // embedding
  k_embed<<<dim3(BS), dim3(128), 0, stream>>>(ids, emb, h32);

  for (int l = 0; l < Ln; ++l) {
    const long dd = (long)l * Dm * Dm;
    conv_launch(stream, Wq + dd, wqb, (long)Dm * Dm);
    conv_launch(stream, Wk + dd, wkb, (long)Dm * Dm);
    conv_launch(stream, Wv + dd, wvb, (long)Dm * Dm);
    conv_launch(stream, Wo + dd, wob, (long)Dm * Dm);
    conv_launch(stream, Wr + dd, wrb, (long)Dm * Dm);
    conv_launch(stream, W1 + (long)l * Dm * FFn, w1b, (long)Dm * FFn);
    conv_launch(stream, W2 + (long)l * FFn * Dm, w2b, (long)FFn * Dm);

    conv_launch(stream, h32, hb, (long)BS * Dm);

    // Q projection -> qc/qp with head biases u,v
    gemm_launch(stream, hb, wqb, tmp32, nullptr, nullptr, BS, Dm, Dm, Dm, Dm, Dm,
                0, 0, 0, 0, 0, 0, 1, 1, 0, 0);
    k_bias_head2<<<dim3(BS * Dm / 256), dim3(256), 0, stream>>>(
        tmp32, uB + (long)l * Dm, vB + (long)l * Dm, qc, qp);
    // K projection
    gemm_launch(stream, hb, wkb, tmp32, nullptr, nullptr, BS, Dm, Dm, Dm, Dm, Dm,
                0, 0, 0, 0, 0, 0, 1, 1, 0, 0);
    conv_launch(stream, tmp32, kb, (long)BS * Dm);
    // V projection
    gemm_launch(stream, hb, wvb, tmp32, nullptr, nullptr, BS, Dm, Dm, Dm, Dm, Dm,
                0, 0, 0, 0, 0, 0, 1, 1, 0, 0);
    conv_launch(stream, tmp32, vb, (long)BS * Dm);
    // Rr = rel_table @ Wr  (only 1025 distinct relative rows needed)
    gemm_launch(stream, relb, wrb, tmp32, nullptr, nullptr, NREL, Dm, Dm, Dm, Dm, Dm,
                0, 0, 0, 0, 0, 0, 1, 1, 0, 0);
    conv_launch(stream, tmp32, rb, (long)NREL * Dm);

    // content[b,h] = (Q+u)[b,h] @ K[b,h]^T   (batched, B^T mode)
    gemm_launch(stream, qc, kb, content, nullptr, nullptr, Sn, Sn, DKn, Dm, Dm, Sn,
                SD, DKn, SD, DKn, (long)Hh * Sn * Sn, (long)Sn * Sn,
                Hh, Bn * Hh, 1, 0);
    // pos_full[b,h] = (Q+v)[b,h] @ Rr_h^T  -> (S, 1025)
    gemm_launch(stream, qp, rb, posf, nullptr, nullptr, Sn, NREL, DKn, Dm, Dm, NREL,
                SD, DKn, 0, DKn, (long)Hh * Sn * NREL, (long)Sn * NREL,
                Hh, Bn * Hh, 1, 0);
    // softmax with relative shift + causal mask -> attn (bf16)
    k_softmax<<<dim3(Bn * Hh * Sn), dim3(128), 0, stream>>>(content, posf, attn_b);
    // out[b,h] = attn[b,h] @ V[b,h]  (batched, row-major B)
    gemm_launch(stream, attn_b, vb, o32, nullptr, nullptr, Sn, DKn, Sn, Sn, Dm, Dm,
                (long)Hh * Sn * Sn, (long)Sn * Sn, SD, DKn, SD, DKn,
                Hh, Bn * Hh, 0, 0);
    // output projection
    conv_launch(stream, o32, ob, (long)BS * Dm);
    gemm_launch(stream, ob, wob, d32, nullptr, nullptr, BS, Dm, Dm, Dm, Dm, Dm,
                0, 0, 0, 0, 0, 0, 1, 1, 0, 0);
    k_addln<<<dim3(BS), dim3(128), 0, stream>>>(h32, d32, g1 + (long)l * Dm, be1 + (long)l * Dm);

    // FFN
    conv_launch(stream, h32, hb, (long)BS * Dm);
    gemm_launch(stream, hb, w1b, nullptr, f1b, b1 + (long)l * FFn, BS, FFn, Dm,
                Dm, FFn, FFn, 0, 0, 0, 0, 0, 0, 1, 1, 0, 1);
    gemm_launch(stream, f1b, w2b, d32, nullptr, b2 + (long)l * Dm, BS, Dm, FFn,
                FFn, Dm, Dm, 0, 0, 0, 0, 0, 0, 1, 1, 0, 0);
    k_addln<<<dim3(BS), dim3(128), 0, stream>>>(h32, d32, g2 + (long)l * Dm, be2 + (long)l * Dm);
  }

  // logits = h @ Wout + bout
  conv_launch(stream, h32, hb, (long)BS * Dm);
  gemm_launch(stream, hb, woutb, out, nullptr, bout, BS, VOCABN, Dm,
              Dm, VOCABN, VOCABN, 0, 0, 0, 0, 0, 0, 1, 1, 0, 0);
}
